// MegaBlocksMoeMLP_45028437131847
// MI455X (gfx1250) — matrix operand order repaired and checked by reading, not compiled, not run on hardware
//
#include <hip/hip_runtime.h>
#include <stdint.h>

// ---------------- problem constants ----------------
#define E_ 16
#define K_ 4
#define H_ 1024
#define F_ 2048
#define T_ 4096
#define N_ (T_*K_)
#define C_ 1024
#define ALPHA_ 1.702f
#define LIMIT_ 7.0f

typedef __attribute__((ext_vector_type(16))) _Float16 v16h;
typedef __attribute__((ext_vector_type(8)))  _Float16 v8h;
typedef __attribute__((ext_vector_type(4)))  _Float16 v4h;
typedef __attribute__((ext_vector_type(8)))  float    v8f;
typedef __attribute__((ext_vector_type(4)))  unsigned int u32x4;
typedef __attribute__((ext_vector_type(8)))  int i32x8;
typedef __attribute__((ext_vector_type(4)))  int i32x4;

#ifdef __HIP_DEVICE_COMPILE__
#  if __has_builtin(__builtin_amdgcn_tensor_load_to_lds)
#    define HAVE_TDM 1
#  else
#    define HAVE_TDM 0
#  endif
#else
#  define HAVE_TDM 0
#endif

// ---------------- WMMA helpers ----------------
__device__ inline v8f wmma16(v16h a, v16h b, v8f c) {
  return __builtin_amdgcn_wmma_f32_16x16x32_f16(false, a, false, b, (short)0, c,
                                                false, false);
}

__device__ inline v16h frag_zero() {
  v16h r;
#pragma unroll
  for (int i = 0; i < 16; ++i) r[i] = (_Float16)0.0f;
  return r;
}

// A 16x32 f16 (ISA 7.12.2): lane<16 holds row, K{0..7,16..23}; lane>=16 K{8..15,24..31}
__device__ inline v16h frag_load_a(const _Float16* p, int lane) {
  const int koff = (lane & 16) ? 8 : 0;
  union { v16h v; v8h h[2]; } r;
  r.h[0] = *(const v8h*)(p + koff);
  r.h[1] = *(const v8h*)(p + 16 + koff);
  return r.v;
}

// B 32x16 f16: lane n holds column n; lanes<16 K{0..7,16..23}, lanes>=16 K{8..15,24..31}
__device__ inline v16h frag_load_b(const _Float16* p, int lane) {
  const int koff = (lane & 16) ? 8 : 0;
  union { v16h v; v8h h[2]; } r;
  r.h[0] = *(const v8h*)(p + koff);
  r.h[1] = *(const v8h*)(p + 16 + koff);
  return r.v;
}

// ---------------- TDM: 128-row x 32-col f16 tile, row stride in elements ----------------
__device__ inline void tdm_load_2d(uint32_t lds_off, const _Float16* gptr,
                                   int stride_elems) {
#if HAVE_TDM
  uint64_t ga = (uint64_t)(uintptr_t)gptr;
  u32x4 g0;
  g0[0] = 1u;                                   // count=1, no gather
  g0[1] = lds_off;                              // lds_addr (bytes)
  g0[2] = (uint32_t)ga;                         // global_addr[95:64]
  g0[3] = ((uint32_t)(ga >> 32) & 0x01FFFFFFu)  // global_addr[120:96]
          | (2u << 30);                         // type=2 (image)
  i32x8 g1;
  g1[0] = (int)(1u << 16);                      // data_size=1 (2 bytes)
  g1[1] = (int)(32u << 16);                     // tensor_dim0 = 32
  g1[2] = (int)(128u << 16);                    // tensor_dim1 = 128
  g1[3] = (int)(32u << 16);                     // tile_dim0 = 32
  g1[4] = 128;                                  // tile_dim1 = 128, tile_dim2 = 0
  g1[5] = stride_elems;                         // tensor_dim0_stride lo32
  g1[6] = 0;
  g1[7] = 0;
  i32x4 z4 = {0, 0, 0, 0};
#if defined(__clang_major__) && (__clang_major__ >= 23)
  i32x8 z8 = {0, 0, 0, 0, 0, 0, 0, 0};
  __builtin_amdgcn_tensor_load_to_lds(g0, g1, z4, z4, z8, 0);
#else
  __builtin_amdgcn_tensor_load_to_lds(g0, g1, z4, z4, 0);
#endif
#else
  (void)lds_off; (void)gptr; (void)stride_elems;
#endif
}

// fallback cooperative copy of a 128x32 f16 panel (256 threads)
__device__ inline void panel_copy(_Float16* dst, const _Float16* src, int stride) {
  const int tid = threadIdx.x;
#pragma unroll
  for (int u = 0; u < 2; ++u) {
    const int unit = tid + u * 256;             // 0..511
    const int row = unit >> 2;
    const int ch  = (unit & 3) * 8;
    *(v8h*)&dst[row * 32 + ch] = *(const v8h*)(src + (size_t)row * stride + ch);
  }
}

// ---------------- x -> f16 ----------------
__global__ void k_cvt_x(const float* __restrict__ x, _Float16* __restrict__ xh) {
  const size_t i = ((size_t)blockIdx.x * blockDim.x + threadIdx.x) * 4;
  float4 v = *(const float4*)(x + i);
  v4h o;
  o[0] = (_Float16)v.x; o[1] = (_Float16)v.y;
  o[2] = (_Float16)v.z; o[3] = (_Float16)v.w;
  *(v4h*)(xh + i) = o;
}

// ---------------- router: logits + top4 + softmax ----------------
__global__ __launch_bounds__(256) void k_router(
    const float* __restrict__ x, const float* __restrict__ rw,
    const float* __restrict__ rb, float* __restrict__ expw,
    float* __restrict__ rsc, int* __restrict__ te) {
  const int tt = threadIdx.x >> 4;  // token in block (16)
  const int e  = threadIdx.x & 15;  // expert
  const int t  = blockIdx.x * 16 + tt;
  const float* xr = x + (size_t)t * H_;
  const float* wr = rw + (size_t)e * H_;
  float acc = rb[e];
  for (int h = 0; h < H_; h += 4) {
    float4 xv = *(const float4*)(xr + h);
    float4 wv = *(const float4*)(wr + h);
    acc = fmaf(xv.x, wv.x, acc);
    acc = fmaf(xv.y, wv.y, acc);
    acc = fmaf(xv.z, wv.z, acc);
    acc = fmaf(xv.w, wv.w, acc);
  }
  __shared__ float lg[16][16];
  __shared__ float sc[16][16];
  lg[tt][e] = acc;
  sc[tt][e] = 0.0f;
  __syncthreads();
  if (e == 0) {
    float v[16];
#pragma unroll
    for (int i = 0; i < 16; ++i) v[i] = lg[tt][i];
    float val[4]; int idx[4];
#pragma unroll
    for (int k = 0; k < 4; ++k) {
      float best = -3.4e38f; int bi = 0;
#pragma unroll
      for (int i = 0; i < 16; ++i) {
        if (v[i] > best) { best = v[i]; bi = i; }
      }
      val[k] = best; idx[k] = bi; v[bi] = -3.4e38f;
    }
    const float m = val[0];
    float wk[4]; float wsum = 0.0f;
#pragma unroll
    for (int k = 0; k < 4; ++k) { wk[k] = __expf(val[k] - m); wsum += wk[k]; }
    const float inv = 1.0f / wsum;
#pragma unroll
    for (int k = 0; k < 4; ++k) {
      wk[k] *= inv;
      expw[(size_t)t * K_ + k] = wk[k];
      te[t * K_ + k] = idx[k];
      sc[tt][idx[k]] = wk[k];
    }
  }
  __syncthreads();
  rsc[(size_t)e * T_ + t] = sc[tt][e];
}

// ---------------- stable counting sort (one block per expert) ----------------
__global__ __launch_bounds__(256) void k_sort(const int* __restrict__ te,
                                              int* __restrict__ tokmap,
                                              int* __restrict__ slotmap) {
  const int e = blockIdx.x;
  const int tid = threadIdx.x;
  for (int s = tid; s < C_; s += 256) tokmap[e * C_ + s] = -1;
  __shared__ int scan[256];
  __shared__ int base;
  if (tid == 0) base = 0;
  __syncthreads();
  for (int c0 = 0; c0 < N_; c0 += 256) {
    const int i = c0 + tid;
    const int pred = (te[i] == e) ? 1 : 0;
    scan[tid] = pred;
    __syncthreads();
    for (int off = 1; off < 256; off <<= 1) {
      int v = (tid >= off) ? scan[tid - off] : 0;
      __syncthreads();
      scan[tid] += v;
      __syncthreads();
    }
    if (pred) {
      const int slot = base + scan[tid] - 1;
      if (slot < C_) {
        tokmap[e * C_ + slot] = i / K_;
        slotmap[i] = e * C_ + slot;
      } else {
        slotmap[i] = -1;
      }
    }
    __syncthreads();
    if (tid == 255) base += scan[255];
    __syncthreads();
  }
}

// ---------------- w1 [E,H,2F] f32 -> w1g/w1u [E,F,H] f16 (de-interleave + transpose) ----------------
__global__ __launch_bounds__(256) void k_trans_w1(const float* __restrict__ w1,
                                                  _Float16* __restrict__ w1g,
                                                  _Float16* __restrict__ w1u) {
  const int e  = blockIdx.z;
  const int h0 = blockIdx.x * 32;
  const int c0 = blockIdx.y * 32;
  const int tx = threadIdx.x & 31;
  const int ty = threadIdx.x >> 5;
  __shared__ float tile[32][33];
  const float* src = w1 + (size_t)e * H_ * (2 * F_);
  for (int hl = ty; hl < 32; hl += 8)
    tile[tx][hl] = src[(size_t)(h0 + hl) * (2 * F_) + c0 + tx];
  __syncthreads();
  for (int cl = ty; cl < 32; cl += 8) {
    const float v = tile[cl][tx];
    const int c = c0 + cl;
    const int f = c >> 1;
    _Float16* dst = (c & 1) ? w1u : w1g;
    dst[((size_t)e * F_ + f) * H_ + h0 + tx] = (_Float16)v;
  }
}

// ---------------- w2 [E,F,H] f32 -> w2t [E,H,F] f16 ----------------
__global__ __launch_bounds__(256) void k_trans_w2(const float* __restrict__ w2,
                                                  _Float16* __restrict__ w2t) {
  const int e  = blockIdx.z;
  const int r0 = blockIdx.x * 32;  // F dim
  const int c0 = blockIdx.y * 32;  // H dim
  const int tx = threadIdx.x & 31;
  const int ty = threadIdx.x >> 5;
  __shared__ float tile[32][33];
  const float* src = w2 + (size_t)e * F_ * H_;
  for (int rl = ty; rl < 32; rl += 8)
    tile[tx][rl] = src[(size_t)(r0 + rl) * H_ + c0 + tx];
  __syncthreads();
  for (int cl = ty; cl < 32; cl += 8)
    w2t[((size_t)e * H_ + c0 + cl) * F_ + r0 + tx] = (_Float16)tile[cl][tx];
}

// ---------------- GEMM1: act[e,c,f] = GLU( gather(xh) @ w1 + b1 ) ----------------
__global__ __launch_bounds__(256) void k_gemm1(
    const _Float16* __restrict__ xh, const int* __restrict__ tokmap,
    const _Float16* __restrict__ w1g, const _Float16* __restrict__ w1u,
    const float* __restrict__ w1b, _Float16* __restrict__ act) {
  const int e  = blockIdx.z;
  const int m0 = blockIdx.x * 64;
  const int n0 = blockIdx.y * 128;
  const int tid  = threadIdx.x;
  const int lane = tid & 31;
  const int wave = tid >> 5;
  const int wm = wave & 1;
  const int wn = wave >> 1;
  const int lr = lane & 15;

  __shared__ _Float16 pg[2][128][32];
  __shared__ _Float16 pu[2][128][32];

  const _Float16* aptr[2];
  bool avalid[2];
#pragma unroll
  for (int i = 0; i < 2; ++i) {
    const int m = m0 + wm * 32 + i * 16 + lr;
    const int tok = tokmap[e * C_ + m];
    avalid[i] = (tok >= 0);
    aptr[i] = xh + (size_t)(tok < 0 ? 0 : tok) * H_;
  }
  const _Float16* gbase = w1g + ((size_t)e * F_ + n0) * H_;
  const _Float16* ubase = w1u + ((size_t)e * F_ + n0) * H_;

  v8f accg[2][2], accu[2][2];
#pragma unroll
  for (int i = 0; i < 2; ++i)
#pragma unroll
    for (int j = 0; j < 2; ++j)
#pragma unroll
      for (int q = 0; q < 8; ++q) { accg[i][j][q] = 0.0f; accu[i][j][q] = 0.0f; }

#if HAVE_TDM
  if (tid < 32) {
    tdm_load_2d((uint32_t)(uintptr_t)&pg[0][0][0], gbase, H_);
    tdm_load_2d((uint32_t)(uintptr_t)&pu[0][0][0], ubase, H_);
  }
#endif
  for (int kt = 0; kt < H_; kt += 32) {
    const int cur = (kt >> 5) & 1;
#if HAVE_TDM
    if (tid < 32) {
      if (kt + 32 < H_) {
        tdm_load_2d((uint32_t)(uintptr_t)&pg[cur ^ 1][0][0], gbase + kt + 32, H_);
        tdm_load_2d((uint32_t)(uintptr_t)&pu[cur ^ 1][0][0], ubase + kt + 32, H_);
        __builtin_amdgcn_s_wait_tensorcnt(2);
      } else {
        __builtin_amdgcn_s_wait_tensorcnt(0);
      }
    }
    __syncthreads();
#else
    __syncthreads();
    panel_copy(&pg[cur][0][0], gbase + kt, H_);
    panel_copy(&pu[cur][0][0], ubase + kt, H_);
    __syncthreads();
#endif
    const v16h a0 = avalid[0] ? frag_load_a(aptr[0] + kt, lane) : frag_zero();
    const v16h a1 = avalid[1] ? frag_load_a(aptr[1] + kt, lane) : frag_zero();
#pragma unroll
    for (int j = 0; j < 2; ++j) {
      const int col = wn * 32 + j * 16 + lr;
      const v16h bg = frag_load_b(&pg[cur][col][0], lane);
      const v16h bu = frag_load_b(&pu[cur][col][0], lane);
      accg[0][j] = wmma16(a0, bg, accg[0][j]);
      accg[1][j] = wmma16(a1, bg, accg[1][j]);
      accu[0][j] = wmma16(a0, bu, accu[0][j]);
      accu[1][j] = wmma16(a1, bu, accu[1][j]);
    }
#if HAVE_TDM
    __syncthreads();
#endif
  }

  const float* bias = w1b + (size_t)e * (2 * F_);
#pragma unroll
  for (int j = 0; j < 2; ++j) {
    const int fcol = n0 + wn * 32 + j * 16 + lr;
    const float bg = bias[2 * fcol];
    const float bu = bias[2 * fcol + 1];
#pragma unroll
    for (int i = 0; i < 2; ++i) {
#pragma unroll
      for (int r = 0; r < 8; ++r) {
        const int m = m0 + wm * 32 + i * 16 + r + ((lane & 16) ? 8 : 0);
        float g = accg[i][j][r] + bg;
        float u = accu[i][j][r] + bu;
        g = fminf(g, LIMIT_);
        u = fminf(fmaxf(u, -LIMIT_), LIMIT_);
        const float sig = 1.0f / (1.0f + __expf(-ALPHA_ * g));
        const float av = (u + 1.0f) * (g * sig);
        act[((size_t)e * C_ + m) * F_ + fcol] = (_Float16)av;
      }
    }
  }
}

// ---------------- GEMM2: y[e,c,h] = act @ w2 + b2 ----------------
__global__ __launch_bounds__(256) void k_gemm2(
    const _Float16* __restrict__ act, const _Float16* __restrict__ w2t,
    const float* __restrict__ w2b, float* __restrict__ ybuf) {
  const int e  = blockIdx.z;
  const int m0 = blockIdx.x * 64;
  const int n0 = blockIdx.y * 128;
  const int tid  = threadIdx.x;
  const int lane = tid & 31;
  const int wave = tid >> 5;
  const int wm = wave & 1;
  const int wn = wave >> 1;
  const int lr = lane & 15;

  __shared__ _Float16 pb[2][128][32];

  const _Float16* aptr[2];
#pragma unroll
  for (int i = 0; i < 2; ++i)
    aptr[i] = act + ((size_t)e * C_ + m0 + wm * 32 + i * 16 + lr) * F_;
  const _Float16* bbase = w2t + ((size_t)e * H_ + n0) * F_;

  v8f acc[2][2];
#pragma unroll
  for (int i = 0; i < 2; ++i)
#pragma unroll
    for (int j = 0; j < 2; ++j)
#pragma unroll
      for (int q = 0; q < 8; ++q) acc[i][j][q] = 0.0f;

#if HAVE_TDM
  if (tid < 32) tdm_load_2d((uint32_t)(uintptr_t)&pb[0][0][0], bbase, F_);
#endif
  for (int kt = 0; kt < F_; kt += 32) {
    const int cur = (kt >> 5) & 1;
#if HAVE_TDM
    if (tid < 32) {
      if (kt + 32 < F_) {
        tdm_load_2d((uint32_t)(uintptr_t)&pb[cur ^ 1][0][0], bbase + kt + 32, F_);
        __builtin_amdgcn_s_wait_tensorcnt(1);
      } else {
        __builtin_amdgcn_s_wait_tensorcnt(0);
      }
    }
    __syncthreads();
#else
    __syncthreads();
    panel_copy(&pb[cur][0][0], bbase + kt, F_);
    __syncthreads();
#endif
    const v16h a0 = frag_load_a(aptr[0] + kt, lane);
    const v16h a1 = frag_load_a(aptr[1] + kt, lane);
#pragma unroll
    for (int j = 0; j < 2; ++j) {
      const int col = wn * 32 + j * 16 + lr;
      const v16h b = frag_load_b(&pb[cur][col][0], lane);
      acc[0][j] = wmma16(a0, b, acc[0][j]);
      acc[1][j] = wmma16(a1, b, acc[1][j]);
    }
#if HAVE_TDM
    __syncthreads();
#endif
  }

#pragma unroll
  for (int j = 0; j < 2; ++j) {
    const int col = n0 + wn * 32 + j * 16 + lr;
    const float b = w2b[(size_t)e * H_ + col];
#pragma unroll
    for (int i = 0; i < 2; ++i) {
#pragma unroll
      for (int r = 0; r < 8; ++r) {
        const int m = m0 + wm * 32 + i * 16 + r + ((lane & 16) ? 8 : 0);
        ybuf[((size_t)e * C_ + m) * H_ + col] = acc[i][j][r] + b;
      }
    }
  }
}

// ---------------- combine: out[t,h] = sum_k w_k * y[slot_k,h] ----------------
__global__ __launch_bounds__(256) void k_combine(const float* __restrict__ ybuf,
                                                 const int* __restrict__ slotmap,
                                                 const float* __restrict__ expw,
                                                 float* __restrict__ out) {
  const int t = blockIdx.x;
  __shared__ int ss[K_];
  __shared__ float sw[K_];
  if (threadIdx.x < K_) {
    ss[threadIdx.x] = slotmap[t * K_ + threadIdx.x];
    sw[threadIdx.x] = expw[(size_t)t * K_ + threadIdx.x];
  }
  __syncthreads();
  for (int h = threadIdx.x; h < H_; h += 256) {
    float s = 0.0f;
#pragma unroll
    for (int k = 0; k < K_; ++k) {
      const int sl = ss[k];
      if (sl >= 0) s += sw[k] * ybuf[(size_t)sl * H_ + h];
    }
    out[(size_t)t * H_ + h] = s;
  }
}

// ---------------- launch ----------------
extern "C" void kernel_launch(void* const* d_in, const int* in_sizes, int n_in,
                              void* d_out, int out_size, void* d_ws, size_t ws_size,
                              hipStream_t stream) {
  (void)in_sizes; (void)n_in; (void)out_size; (void)ws_size;
  const float* x   = (const float*)d_in[0];
  const float* rw  = (const float*)d_in[1];
  const float* rb  = (const float*)d_in[2];
  const float* w1  = (const float*)d_in[3];
  const float* w2  = (const float*)d_in[4];
  const float* w1b = (const float*)d_in[5];
  const float* w2b = (const float*)d_in[6];

  float* out  = (float*)d_out;
  float* expw = out + (size_t)T_ * H_;
  float* rsc  = expw + (size_t)T_ * K_;

  char* p = (char*)d_ws;
  _Float16* xh  = (_Float16*)p; p += (size_t)T_ * H_ * 2;
  _Float16* w1g = (_Float16*)p; p += (size_t)E_ * F_ * H_ * 2;
  _Float16* w1u = (_Float16*)p; p += (size_t)E_ * F_ * H_ * 2;
  _Float16* w2t = (_Float16*)p; p += (size_t)E_ * H_ * F_ * 2;
  _Float16* act = (_Float16*)p; p += (size_t)E_ * C_ * F_ * 2;
  float*    ybuf= (float*)p;    p += (size_t)E_ * C_ * H_ * 4;
  int* te      = (int*)p; p += (size_t)N_ * 4;
  int* slotmap = (int*)p; p += (size_t)N_ * 4;
  int* tokmap  = (int*)p; p += (size_t)E_ * C_ * 4;

  k_cvt_x<<<(T_ * H_) / 4 / 256, 256, 0, stream>>>(x, xh);
  k_router<<<T_ / 16, 256, 0, stream>>>(x, rw, rb, expw, rsc, te);
  k_sort<<<E_, 256, 0, stream>>>(te, tokmap, slotmap);
  k_trans_w1<<<dim3(H_ / 32, (2 * F_) / 32, E_), 256, 0, stream>>>(w1, w1g, w1u);
  k_trans_w2<<<dim3(F_ / 32, H_ / 32, E_), 256, 0, stream>>>(w2, w2t);
  k_gemm1<<<dim3(C_ / 64, F_ / 128, E_), 256, 0, stream>>>(xh, tokmap, w1g, w1u,
                                                           w1b, act);
  k_gemm2<<<dim3(C_ / 64, H_ / 128, E_), 256, 0, stream>>>(act, w2t, w2b, ybuf);
  k_combine<<<T_, 256, 0, stream>>>(ybuf, slotmap, expw, out);
}
